// LVGNN_35373350650220
// MI455X (gfx1250) — compile-verified, matrix operand order checked by reading, not run on hardware
//
#include <hip/hip_runtime.h>

// ---------------------------------------------------------------------------
// Types for CDNA5 (gfx1250) WMMA: wave32, 16x16x32 bf16 -> f32 accumulate
// ---------------------------------------------------------------------------
typedef __bf16 bf16;
typedef __bf16 v8bf  __attribute__((ext_vector_type(8)));
typedef __bf16 v16bf __attribute__((ext_vector_type(16)));
typedef float  v8f   __attribute__((ext_vector_type(8)));

#define HID       128
#define IN_NODE   16
#define IN_EDGE   16
#define KE        160      // 128 (h) + 16 (edge_attr) + 16 zero pad -> 5 k-steps
#define KSTEPS_E  5
#define KSTEPS_H  4        // K = 128 -> 4 k-steps of 32

// A-fragment per lane for V_WMMA_F32_16X16X32_BF16 (ISA 7.12.2, 16-bit A 16x32):
// lane l holds row M = l%16; VGPR0..3 = K in [kbase + (l/16)*8, +8),
// VGPR4..7 = K in [kbase + 16 + (l/16)*8, +8).  Two 16B contiguous LDS reads.
static __device__ __forceinline__ v16bf load_a_frag(const bf16* rowptr, int ks, int sel) {
    const bf16* ap = rowptr + ks * 32 + sel * 8;
    v8bf lo = *(const v8bf*)ap;
    v8bf hi = *(const v8bf*)(ap + 16);
    v16bf a;
#pragma unroll
    for (int i = 0; i < 8; ++i) { a[i] = lo[i]; a[8 + i] = hi[i]; }
    return a;
}

static __device__ __forceinline__ v8f wmma_bf16(v16bf a, v16bf b, v8f c) {
    // (neg_a, A, neg_b, B, c_mod, C, reuse_a, reuse_b)
    return __builtin_amdgcn_wmma_f32_16x16x32_bf16(false, a, false, b, (short)0, c,
                                                   false, false);
}

// ---------------------------------------------------------------------------
// Weight pre-swizzle: f32 [Ksrc, Nsrc] row-major -> bf16 fragment-major layout
// so a wave's B fragment for (ntile, kstep) is 32 lanes x 16 contiguous bf16.
// Lane l of fragment (nt, ks): col N = nt*16 + l%16,
//   elems 0..7 : K = ks*32 + (l/16)*8 + i
//   elems 8..15: K = ks*32 + 16 + (l/16)*8 + (i-8)
// Zero-pad K >= Ksrc and N >= Nsrc.
// ---------------------------------------------------------------------------
__global__ void swizzle_weight_kernel(const float* __restrict__ W, bf16* __restrict__ dst,
                                      int Ksrc, int Nsrc, int ksteps) {
    int bid  = blockIdx.x;
    int ks   = bid % ksteps;
    int nt   = bid / ksteps;
    int lane = threadIdx.x;          // 0..31
    int n    = nt * 16 + (lane & 15);
    int sel  = lane >> 4;
    bf16* p  = dst + ((size_t)(nt * ksteps + ks) * 32 + lane) * 16;
#pragma unroll
    for (int i = 0; i < 16; ++i) {
        int k = ks * 32 + ((i < 8) ? (sel * 8 + i) : (16 + sel * 8 + (i - 8)));
        float v = (k < Ksrc && n < Nsrc) ? W[(size_t)k * Nsrc + n] : 0.0f;
        p[i] = (bf16)v;
    }
}

__global__ void zero_f32_kernel(float* __restrict__ p, size_t n) {
    size_t i = (size_t)blockIdx.x * blockDim.x + threadIdx.x;
    size_t stride = (size_t)gridDim.x * blockDim.x;
    for (; i < n; i += stride) p[i] = 0.0f;
}

// ---------------------------------------------------------------------------
// Embedding: h0 = relu(x @ Wemb1 + b1) @ Wemb2 + b2   (bf16 out)
// One block = 16 node rows, 8 waves (one 16-col slab each).
// ---------------------------------------------------------------------------
__global__ void embed_kernel(const float* __restrict__ x,
                             const bf16* __restrict__ W1sw, const float* __restrict__ b1,
                             const bf16* __restrict__ W2sw, const float* __restrict__ b2,
                             bf16* __restrict__ hout, int Nn) {
    __shared__ bf16 As[16][32];
    __shared__ bf16 Ts[16][HID];
    const int tid  = threadIdx.x;
    const int lane = tid & 31;
    const int wave = tid >> 5;                 // n-tile id
    const int base = blockIdx.x * 16;
    {   // load x tile (16x16 f32 -> bf16), pad K 16..31 with zeros
        int r = tid >> 4, c = tid & 15;
        float xv = (base + r < Nn) ? x[(size_t)(base + r) * IN_NODE + c] : 0.0f;
        As[r][c]      = (bf16)xv;
        As[r][16 + c] = (bf16)0.0f;
    }
    __syncthreads();
    const int sel = lane >> 4, row = lane & 15;
    const int col = wave * 16 + (lane & 15);
    // GEMM1: K padded to 32 -> one WMMA
    v8f c1 = {};
    {
        v16bf a = load_a_frag(&As[row][0], 0, sel);
        v16bf b = ((const v16bf*)W1sw)[(size_t)wave * 32 + lane];
        c1 = wmma_bf16(a, b, c1);
    }
    float bias1 = b1[col];
#pragma unroll
    for (int i = 0; i < 8; ++i) {
        int m = i + (lane >> 4) * 8;
        float v = c1[i] + bias1;
        Ts[m][col] = (bf16)(v > 0.0f ? v : 0.0f);
    }
    __syncthreads();
    // GEMM2: K = 128 -> 4 WMMAs
    v8f c2 = {};
#pragma unroll
    for (int ks = 0; ks < KSTEPS_H; ++ks) {
        v16bf a = load_a_frag(&Ts[row][0], ks, sel);
        v16bf b = ((const v16bf*)W2sw)[(size_t)(wave * KSTEPS_H + ks) * 32 + lane];
        c2 = wmma_bf16(a, b, c2);
    }
    float bias2 = b2[col];
#pragma unroll
    for (int i = 0; i < 8; ++i) {
        int m = i + (lane >> 4) * 8;
        int node = base + m;
        if (node < Nn) hout[(size_t)node * HID + col] = (bf16)(c2[i] + bias2);
    }
}

// ---------------------------------------------------------------------------
// Edge message + scatter:  m = [h[src] | edge_attr] @ Wm + bm ; agg[dst] += m
// Block = 256 thr = 8 waves; grid-stride over 16-edge tiles; B frags cached in
// registers for the whole loop (5 v16bf per wave). Scatter via f32 atomics.
// ---------------------------------------------------------------------------
__global__ void edge_msg_kernel(const bf16* __restrict__ h,
                                const float* __restrict__ edge_attr,
                                const int* __restrict__ src, const int* __restrict__ dst,
                                const bf16* __restrict__ Wmsw, const float* __restrict__ bm,
                                float* __restrict__ agg, int E) {
    __shared__ bf16 As[16][KE];
    __shared__ int  sdst[16];
    const int tid  = threadIdx.x;
    const int lane = tid & 31;
    const int wave = tid >> 5;                 // n-tile id (0..7)
    const int ntiles = (E + 15) >> 4;

    v16bf Bf[KSTEPS_E];
#pragma unroll
    for (int ks = 0; ks < KSTEPS_E; ++ks)
        Bf[ks] = ((const v16bf*)Wmsw)[(size_t)(wave * KSTEPS_E + ks) * 32 + lane];

    const int col  = wave * 16 + (lane & 15);
    const float bias = bm[col];
    const int sel = lane >> 4, row = lane & 15;

    for (int tile = blockIdx.x; tile < ntiles; tile += gridDim.x) {
        const int base = tile * 16;
        {   // cooperative gather: thread (r, p): r = edge row, p = 8-elem chunk
            int r = tid >> 4, p = tid & 15;
            int e = base + r;
            bool ok = (e < E);
            int s = ok ? src[e] : 0;
            if (p == 0) sdst[r] = ok ? dst[e] : -1;
            v8bf v;
#pragma unroll
            for (int i = 0; i < 8; ++i) v[i] = (bf16)0.0f;
            if (ok) v = *(const v8bf*)(h + (size_t)s * HID + p * 8);
            *(v8bf*)&As[r][p * 8] = v;                                   // h part
            float ea = ok ? edge_attr[(size_t)e * IN_EDGE + p] : 0.0f;
            As[r][HID + p]      = (bf16)ea;                              // edge feat
            As[r][HID + 16 + p] = (bf16)0.0f;                            // K pad
        }
        __syncthreads();

        v8f c = {};
#pragma unroll
        for (int ks = 0; ks < KSTEPS_E; ++ks) {
            v16bf a = load_a_frag(&As[row][0], ks, sel);
            c = wmma_bf16(a, Bf[ks], c);
        }

        // scatter-add: VGPR i of lane -> edge row (i + 8*(lane/16)), column col
#pragma unroll
        for (int i = 0; i < 8; ++i) {
            int m = i + (lane >> 4) * 8;
            int d = sdst[m];
            if (d >= 0) atomicAdd(&agg[(size_t)d * HID + col], c[i] + bias);
        }
        __syncthreads();
    }
}

// ---------------------------------------------------------------------------
// Node update: h' = relu(h @ Ws + bs + agg)   (bf16 out)
// ---------------------------------------------------------------------------
__global__ void node_update_kernel(const bf16* __restrict__ h,
                                   const bf16* __restrict__ Wssw, const float* __restrict__ bs,
                                   const float* __restrict__ agg, bf16* __restrict__ hout,
                                   int Nn) {
    __shared__ bf16 As[16][HID];
    const int tid  = threadIdx.x;
    const int lane = tid & 31;
    const int wave = tid >> 5;
    const int base = blockIdx.x * 16;
    {
        int r = tid >> 4, p = tid & 15;
        v8bf v;
#pragma unroll
        for (int i = 0; i < 8; ++i) v[i] = (bf16)0.0f;
        if (base + r < Nn) v = *(const v8bf*)(h + (size_t)(base + r) * HID + p * 8);
        *(v8bf*)&As[r][p * 8] = v;
    }
    __syncthreads();
    const int sel = lane >> 4, row = lane & 15;
    v8f c = {};
#pragma unroll
    for (int ks = 0; ks < KSTEPS_H; ++ks) {
        v16bf a = load_a_frag(&As[row][0], ks, sel);
        v16bf b = ((const v16bf*)Wssw)[(size_t)(wave * KSTEPS_H + ks) * 32 + lane];
        c = wmma_bf16(a, b, c);
    }
    const int col = wave * 16 + (lane & 15);
    const float bias = bs[col];
#pragma unroll
    for (int i = 0; i < 8; ++i) {
        int m = i + (lane >> 4) * 8;
        int node = base + m;
        if (node < Nn) {
            float v = c[i] + bias + agg[(size_t)node * HID + col];
            hout[(size_t)node * HID + col] = (bf16)(v > 0.0f ? v : 0.0f);
        }
    }
}

// ---------------------------------------------------------------------------
// Head: out = relu(h @ Wh1 + b1) @ Wh2 + b2   (f32 out, OUT=3, padded to 16)
// ---------------------------------------------------------------------------
__global__ void head_kernel(const bf16* __restrict__ h,
                            const bf16* __restrict__ W1sw, const float* __restrict__ b1,
                            const bf16* __restrict__ W2sw, const float* __restrict__ b2,
                            float* __restrict__ out, int Nn, int OutDim) {
    __shared__ bf16 As[16][HID];
    __shared__ bf16 Ts[16][HID];
    const int tid  = threadIdx.x;
    const int lane = tid & 31;
    const int wave = tid >> 5;
    const int base = blockIdx.x * 16;
    {
        int r = tid >> 4, p = tid & 15;
        v8bf v;
#pragma unroll
        for (int i = 0; i < 8; ++i) v[i] = (bf16)0.0f;
        if (base + r < Nn) v = *(const v8bf*)(h + (size_t)(base + r) * HID + p * 8);
        *(v8bf*)&As[r][p * 8] = v;
    }
    __syncthreads();
    const int sel = lane >> 4, row = lane & 15;
    const int col = wave * 16 + (lane & 15);
    v8f c1 = {};
#pragma unroll
    for (int ks = 0; ks < KSTEPS_H; ++ks) {
        v16bf a = load_a_frag(&As[row][0], ks, sel);
        v16bf b = ((const v16bf*)W1sw)[(size_t)(wave * KSTEPS_H + ks) * 32 + lane];
        c1 = wmma_bf16(a, b, c1);
    }
    float bias1 = b1[col];
#pragma unroll
    for (int i = 0; i < 8; ++i) {
        int m = i + (lane >> 4) * 8;
        float v = c1[i] + bias1;
        Ts[m][col] = (bf16)(v > 0.0f ? v : 0.0f);
    }
    __syncthreads();
    if (wave == 0) {     // single n-tile of 16 cols (only first OutDim valid)
        v8f c2 = {};
#pragma unroll
        for (int ks = 0; ks < KSTEPS_H; ++ks) {
            v16bf a = load_a_frag(&Ts[row][0], ks, sel);
            v16bf b = ((const v16bf*)W2sw)[(size_t)ks * 32 + lane];
            c2 = wmma_bf16(a, b, c2);
        }
        int oc = lane & 15;
        if (oc < OutDim) {
            float bias2 = b2[oc];
#pragma unroll
            for (int i = 0; i < 8; ++i) {
                int m = i + (lane >> 4) * 8;
                int node = base + m;
                if (node < Nn) out[(size_t)node * OutDim + oc] = c2[i] + bias2;
            }
        }
    }
}

// ---------------------------------------------------------------------------
extern "C" void kernel_launch(void* const* d_in, const int* in_sizes, int n_in,
                              void* d_out, int out_size, void* d_ws, size_t ws_size,
                              hipStream_t stream) {
    const float* x      = (const float*)d_in[0];
    const int*   eidx   = (const int*)d_in[1];
    const float* eattr  = (const float*)d_in[2];
    const float* Wemb1  = (const float*)d_in[3];
    const float* bemb1  = (const float*)d_in[4];
    const float* Wemb2  = (const float*)d_in[5];
    const float* bemb2  = (const float*)d_in[6];
    const float* Ws[3]  = {(const float*)d_in[7],  (const float*)d_in[11], (const float*)d_in[15]};
    const float* bs[3]  = {(const float*)d_in[8],  (const float*)d_in[12], (const float*)d_in[16]};
    const float* Wm[3]  = {(const float*)d_in[9],  (const float*)d_in[13], (const float*)d_in[17]};
    const float* bm[3]  = {(const float*)d_in[10], (const float*)d_in[14], (const float*)d_in[18]};
    const float* Whead1 = (const float*)d_in[19];
    const float* bhead1 = (const float*)d_in[20];
    const float* Whead2 = (const float*)d_in[21];
    const float* bhead2 = (const float*)d_in[22];

    const int Nn = in_sizes[0] / IN_NODE;      // 50000
    const int E  = in_sizes[2] / IN_EDGE;      // 1600000
    const int OutDim = out_size / Nn;          // 3
    const int* src = eidx;
    const int* dst = eidx + E;

    // ---- workspace carve-out (256B aligned) ----
    char* ws = (char*)d_ws;
    auto take = [&](size_t bytes) {
        char* p = ws;
        ws += (bytes + 255) & ~(size_t)255;
        return p;
    };
    const size_t hElems = (size_t)Nn * HID;
    bf16*  hA    = (bf16*)take(hElems * sizeof(bf16));
    bf16*  hB    = (bf16*)take(hElems * sizeof(bf16));
    float* agg   = (float*)take(hElems * sizeof(float));
    bf16*  w_emb1 = (bf16*)take((size_t)8 * 1 * 512 * sizeof(bf16));
    bf16*  w_emb2 = (bf16*)take((size_t)8 * 4 * 512 * sizeof(bf16));
    bf16*  w_m[3], *w_s[3];
    for (int i = 0; i < 3; ++i) w_m[i] = (bf16*)take((size_t)8 * 5 * 512 * sizeof(bf16));
    for (int i = 0; i < 3; ++i) w_s[i] = (bf16*)take((size_t)8 * 4 * 512 * sizeof(bf16));
    bf16*  w_h1  = (bf16*)take((size_t)8 * 4 * 512 * sizeof(bf16));
    bf16*  w_h2  = (bf16*)take((size_t)1 * 4 * 512 * sizeof(bf16));

    // ---- weight prep (f32 -> bf16, fragment-major, zero-padded) ----
    swizzle_weight_kernel<<<8 * 1, 32, 0, stream>>>(Wemb1, w_emb1, IN_NODE, HID, 1);
    swizzle_weight_kernel<<<8 * 4, 32, 0, stream>>>(Wemb2, w_emb2, HID, HID, KSTEPS_H);
    for (int i = 0; i < 3; ++i) {
        swizzle_weight_kernel<<<8 * 5, 32, 0, stream>>>(Wm[i], w_m[i], HID + IN_EDGE, HID, KSTEPS_E);
        swizzle_weight_kernel<<<8 * 4, 32, 0, stream>>>(Ws[i], w_s[i], HID, HID, KSTEPS_H);
    }
    swizzle_weight_kernel<<<8 * 4, 32, 0, stream>>>(Whead1, w_h1, HID, HID, KSTEPS_H);
    swizzle_weight_kernel<<<1 * 4, 32, 0, stream>>>(Whead2, w_h2, HID, OutDim, KSTEPS_H);

    const int nodeBlocks = (Nn + 15) / 16;

    // ---- embedding ----
    embed_kernel<<<nodeBlocks, 256, 0, stream>>>(x, w_emb1, bemb1, w_emb2, bemb2, hA, Nn);

    // ---- 3 message-passing layers ----
    bf16* hin = hA;
    bf16* hout = hB;
    for (int l = 0; l < 3; ++l) {
        zero_f32_kernel<<<2048, 256, 0, stream>>>(agg, hElems);
        edge_msg_kernel<<<3072, 256, 0, stream>>>(hin, eattr, src, dst, w_m[l], bm[l], agg, E);
        node_update_kernel<<<nodeBlocks, 256, 0, stream>>>(hin, w_s[l], bs[l], agg, hout, Nn);
        bf16* t = hin; hin = hout; hout = t;
    }

    // ---- head ----
    head_kernel<<<nodeBlocks, 256, 0, stream>>>(hin, w_h1, bhead1, w_h2, bhead2,
                                                (float*)d_out, Nn, OutDim);
}
